// JTNNEncoder_25735444037937
// MI455X (gfx1250) — matrix operand order repaired and checked by reading, not compiled, not run on hardware
//
#include <hip/hip_runtime.h>
#include <hip/hip_bf16.h>

typedef _Float16 half_t;
typedef __attribute__((ext_vector_type(16))) _Float16 v16h;
typedef __attribute__((ext_vector_type(8)))  _Float16 v8h;
typedef __attribute__((ext_vector_type(8)))  float    v8f;
typedef __attribute__((ext_vector_type(4)))  int      v4i_;

static constexpr int H      = 256;
static constexpr int MMSG   = 131072;
static constexpr int NNODE  = 65536;
static constexpr int NTREE  = 2048;
static constexpr int MAXNEI = 5;

// ---------------------------------------------------------------------------
// Async global->LDS staging (CDNA5 GLOBAL_LOAD_ASYNC_TO_LDS_B128, ASYNCcnt),
// with a synchronous uint4 fallback if the builtin is unavailable.
// Probe-derived signature: (v4i AS(1)*, v4i AS(3)*, imm offset, imm cpol).
// ---------------------------------------------------------------------------
#if defined(__gfx1250__) && __has_builtin(__builtin_amdgcn_global_load_async_to_lds_b128)
#define HAVE_ASYNC_LDS 1
#endif

__device__ __forceinline__ void stage16(const void* g, void* l) {
#ifdef HAVE_ASYNC_LDS
    __builtin_amdgcn_global_load_async_to_lds_b128(
        (__attribute__((address_space(1))) v4i_*)(void*)g,
        (__attribute__((address_space(3))) v4i_*)l, 0, 0);
#else
    *(uint4*)l = *(const uint4*)g;
#endif
}

__device__ __forceinline__ void stage_wait() {
#ifdef HAVE_ASYNC_LDS
#if __has_builtin(__builtin_amdgcn_s_wait_asynccnt)
    __builtin_amdgcn_s_wait_asynccnt(0);
#else
    asm volatile("s_wait_asynccnt 0x0" ::: "memory");
#endif
#endif
}

__device__ __forceinline__ float sigmoidf_(float v) {
    return 1.0f / (1.0f + __expf(-v));
}

// ---------------------------------------------------------------------------
// Reorder a [256(k) x 256(n)] f32 weight into B-fragment order (f16):
//   dst[(kt*16+nt)*32 + lane][s] = W[32*kt + (lane<16?0:8) + (s<8?s:s+8)][16*nt + lane%16]
// CDNA5 16-bit B 32x16 layout: lanes 0-15 hold K=0..7,16..23, lanes 16-31 K=8..15,24..31,
// N = lane%16 (matches A-fragment K order).
// ---------------------------------------------------------------------------
__global__ void reorder_w(const float* __restrict__ W, half_t* __restrict__ dst) {
    int tile = blockIdx.x;          // 0..127 = kt*16+nt
    int l    = threadIdx.x;         // 0..31
    int kt = tile >> 4, nt = tile & 15;
    int kg  = (l < 16) ? 0 : 8;
    int col = nt * 16 + (l & 15);
    half_t* d = dst + ((size_t)(tile * 32 + l)) * 16;
#pragma unroll
    for (int s = 0; s < 16; ++s)
        d[s] = (_Float16)W[(size_t)(kt * 32 + kg + (s < 8 ? s : s + 8)) * H + col];
}

// ---------------------------------------------------------------------------
// Per-wave A-fragment preload (CDNA5 16-bit A 16x32 layout): lanes 0-15 hold
// K slots {0..7}->{K..K+7}, {8..15}->{K+16..K+23}; lanes 16-31 shifted by 8.
// ---------------------------------------------------------------------------
__device__ __forceinline__ void load_afrags(const half_t* __restrict__ A,
                                            int rowA, int koff, v16h afrag[8]) {
    const half_t* Arow = A + (size_t)rowA * H;
#pragma unroll
    for (int kt = 0; kt < 8; ++kt) {
        v8h lo = *(const v8h*)(Arow + kt * 32 + koff);
        v8h hi = *(const v8h*)(Arow + kt * 32 + koff + 16);
        afrag[kt] = __builtin_shufflevector(lo, hi,
            0,1,2,3,4,5,6,7,8,9,10,11,12,13,14,15);
    }
}

// ---------------------------------------------------------------------------
// C[M,256] = A_f16[M,256] @ W(+bias), f32 out. Dual WMMA chains + pipelined
// B-fragment LDS loads.
// ---------------------------------------------------------------------------
__global__ void __launch_bounds__(256) gemm_k256(const half_t* __restrict__ A,
                                                 const half_t* __restrict__ Wrf,
                                                 const float* __restrict__ bias,
                                                 float* __restrict__ C) {
    __shared__ __align__(16) half_t sW[8][8][32][16];   // [kt][ntLocal][lane][16] = 64KB

    const int tid  = threadIdx.x;
    const int wave = tid >> 5;
    const int lane = tid & 31;
    const int m0   = blockIdx.x * 128 + wave * 16;

    v16h afrag[8];
    load_afrags(A, m0 + (lane & 15), (lane < 16) ? 0 : 8, afrag);

    const int colBase = lane & 15;
    const int rowbase = m0 + ((lane < 16) ? 0 : 8);

    for (int hlf = 0; hlf < 2; ++hlf) {
        __syncthreads();
        for (int i = tid; i < 4096; i += 256) {               // 64KB stage
            int tileL = i >> 6;                                // kt*8 + ntl
            int rest  = i & 63;
            int kt = tileL >> 3, ntl = tileL & 7;
            int nt = hlf * 8 + ntl;
            stage16(((const uint4*)Wrf) + (size_t)(kt * 16 + nt) * 64 + rest,
                    ((uint4*)sW) + i);
        }
        stage_wait();
        __syncthreads();

        for (int ntl = 0; ntl < 8; ++ntl) {
            int col = (hlf * 8 + ntl) * 16 + colBase;
            const half_t* bp = (const half_t*)sW + ntl * 512 + lane * 16;  // kt stride 4096
            v16h b0 = *(const v16h*)(bp);
            v16h b1 = *(const v16h*)(bp + 4096);
            float b = bias ? bias[col] : 0.0f;
            v8f acc0 = {b, b, b, b, b, b, b, b};
            v8f acc1 = {0.f, 0.f, 0.f, 0.f, 0.f, 0.f, 0.f, 0.f};
#pragma unroll
            for (int kt = 0; kt < 8; kt += 2) {
                v16h p0 = (kt < 6) ? *(const v16h*)(bp + (kt + 2) * 4096) : b0;
                acc0 = __builtin_amdgcn_wmma_f32_16x16x32_f16(
                         false, afrag[kt],     false, b0, (short)0, acc0, false, false);
                v16h p1 = (kt < 6) ? *(const v16h*)(bp + (kt + 3) * 4096) : b1;
                acc1 = __builtin_amdgcn_wmma_f32_16x16x32_f16(
                         false, afrag[kt + 1], false, b1, (short)0, acc1, false, false);
                b0 = p0; b1 = p1;
            }
            v8f acc = acc0 + acc1;
            float* Cc = C + (size_t)rowbase * H + col;
#pragma unroll
            for (int v = 0; v < 8; ++v)
                Cc[(size_t)v * H] = acc[v];
        }
    }
}

// ---------------------------------------------------------------------------
// Fused per-depth gate GEMMs + GRU combine:
//   zlin = sumh16 @ Wz_bot ; plin = sumgh16 @ Wh_bot (both in registers)
//   z = sigmoid(xWz + zlin); p = tanh(xWh + plin)
//   h = ((1-z)*sum_h + z*p) * mask  -> h32, h16
// ---------------------------------------------------------------------------
__global__ void __launch_bounds__(256) gemm2_gate_combine(
        const half_t* __restrict__ As, const half_t* __restrict__ Ag,
        const half_t* __restrict__ Wzb, const half_t* __restrict__ Whb,
        const float* __restrict__ xWz, const float* __restrict__ xWh,
        const float* __restrict__ sumh32,
        float* __restrict__ h32, half_t* __restrict__ h16) {
    __shared__ __align__(16) half_t sZ[8][4][32][16];   // 32KB  [kt][ntl][lane][16]
    __shared__ __align__(16) half_t sH[8][4][32][16];   // 32KB

    const int tid  = threadIdx.x;
    const int wave = tid >> 5;
    const int lane = tid & 31;
    const int m0   = blockIdx.x * 128 + wave * 16;

    v16h afS[8], afG[8];
    load_afrags(As, m0 + (lane & 15), (lane < 16) ? 0 : 8, afS);
    load_afrags(Ag, m0 + (lane & 15), (lane < 16) ? 0 : 8, afG);

    const int colBase = lane & 15;
    const int rowbase = m0 + ((lane < 16) ? 0 : 8);

    for (int q = 0; q < 4; ++q) {                         // 4 groups of 4 N-tiles
        __syncthreads();
        for (int i = tid; i < 2048; i += 256) {           // 32KB per matrix
            int tileL = i >> 6;                            // kt*4 + ntl
            int rest  = i & 63;
            int kt = tileL >> 2, ntl = tileL & 3;
            int nt = q * 4 + ntl;
            size_t src = (size_t)(kt * 16 + nt) * 64 + rest;
            stage16(((const uint4*)Wzb) + src, ((uint4*)sZ) + i);
            stage16(((const uint4*)Whb) + src, ((uint4*)sH) + i);
        }
        stage_wait();
        __syncthreads();

        for (int ntl = 0; ntl < 4; ++ntl) {
            int col = (q * 4 + ntl) * 16 + colBase;
            const half_t* bpz = (const half_t*)sZ + ntl * 512 + lane * 16; // kt stride 2048
            const half_t* bph = (const half_t*)sH + ntl * 512 + lane * 16;
            v16h bz = *(const v16h*)(bpz);
            v16h bh = *(const v16h*)(bph);
            v8f accZ = {0.f, 0.f, 0.f, 0.f, 0.f, 0.f, 0.f, 0.f};
            v8f accH = accZ;
#pragma unroll
            for (int kt = 0; kt < 8; ++kt) {
                v16h pz = (kt < 7) ? *(const v16h*)(bpz + (kt + 1) * 2048) : bz;
                accZ = __builtin_amdgcn_wmma_f32_16x16x32_f16(
                         false, afS[kt], false, bz, (short)0, accZ, false, false);
                v16h ph = (kt < 7) ? *(const v16h*)(bph + (kt + 1) * 2048) : bh;
                accH = __builtin_amdgcn_wmma_f32_16x16x32_f16(
                         false, afG[kt], false, bh, (short)0, accH, false, false);
                bz = pz; bh = ph;
            }
#pragma unroll
            for (int v = 0; v < 8; ++v) {
                int row = rowbase + v;
                size_t i = (size_t)row * H + col;
                float z  = sigmoidf_(xWz[i] + accZ[v]);
                float p  = tanhf(xWh[i] + accH[v]);
                float hv = (1.0f - z) * sumh32[i] + z * p;
                if (row == 0) hv = 0.0f;                  // message 0 is padding
                h32[i] = hv;
                h16[i] = (_Float16)hv;
            }
        }
    }
}

// ---------------------------------------------------------------------------
// Fused readout GEMM: nv = relu(xo + mess_nei@out_bot); atomic segment-sum.
// ---------------------------------------------------------------------------
__global__ void __launch_bounds__(256) gemm_out_accum(
        const half_t* __restrict__ A, const half_t* __restrict__ Wrf,
        const float* __restrict__ xo, const int* __restrict__ scope,
        float* __restrict__ sums) {
    __shared__ __align__(16) half_t sW[8][8][32][16];

    const int tid  = threadIdx.x;
    const int wave = tid >> 5;
    const int lane = tid & 31;
    const int m0   = blockIdx.x * 128 + wave * 16;

    v16h afrag[8];
    load_afrags(A, m0 + (lane & 15), (lane < 16) ? 0 : 8, afrag);

    const int colBase = lane & 15;
    const int rowbase = m0 + ((lane < 16) ? 0 : 8);

    for (int hlf = 0; hlf < 2; ++hlf) {
        __syncthreads();
        for (int i = tid; i < 4096; i += 256) {
            int tileL = i >> 6;
            int rest  = i & 63;
            int kt = tileL >> 3, ntl = tileL & 7;
            int nt = hlf * 8 + ntl;
            stage16(((const uint4*)Wrf) + (size_t)(kt * 16 + nt) * 64 + rest,
                    ((uint4*)sW) + i);
        }
        stage_wait();
        __syncthreads();

        for (int ntl = 0; ntl < 8; ++ntl) {
            int col = (hlf * 8 + ntl) * 16 + colBase;
            const half_t* bp = (const half_t*)sW + ntl * 512 + lane * 16;
            v16h b0 = *(const v16h*)(bp);
            v16h b1 = *(const v16h*)(bp + 4096);
            v8f acc0 = {0.f, 0.f, 0.f, 0.f, 0.f, 0.f, 0.f, 0.f};
            v8f acc1 = acc0;
#pragma unroll
            for (int kt = 0; kt < 8; kt += 2) {
                v16h p0 = (kt < 6) ? *(const v16h*)(bp + (kt + 2) * 4096) : b0;
                acc0 = __builtin_amdgcn_wmma_f32_16x16x32_f16(
                         false, afrag[kt],     false, b0, (short)0, acc0, false, false);
                v16h p1 = (kt < 6) ? *(const v16h*)(bp + (kt + 3) * 4096) : b1;
                acc1 = __builtin_amdgcn_wmma_f32_16x16x32_f16(
                         false, afrag[kt + 1], false, b1, (short)0, acc1, false, false);
                b0 = p0; b1 = p1;
            }
            v8f acc = acc0 + acc1;
#pragma unroll
            for (int v = 0; v < 8; ++v) {
                int row = rowbase + v;
                size_t i = (size_t)row * H + col;
                float nv = xo[i] + acc[v];
                nv = nv > 0.0f ? nv : 0.0f;
                atomicAdd(&sums[(size_t)scope[row] * H + col], nv);
            }
        }
    }
}

// ---------------------------------------------------------------------------
// Elementwise / gather kernels (blockDim = 256 = H)
// ---------------------------------------------------------------------------
__global__ void zero_f(float* __restrict__ p, long n) {
    long i = (long)blockIdx.x * blockDim.x + threadIdx.x;
    if (i < n) p[i] = 0.0f;
}

__global__ void embed_node(const int* __restrict__ fnode, const float* __restrict__ emb,
                           half_t* __restrict__ xn16) {
    int n = blockIdx.x, j = threadIdx.x;
    xn16[(size_t)n * H + j] = (_Float16)emb[(size_t)fnode[n] * H + j];
}

__global__ void embed_mess(const int* __restrict__ fmess, const half_t* __restrict__ xn16,
                           half_t* __restrict__ x16) {
    int m = blockIdx.x, j = threadIdx.x;
    x16[(size_t)m * H + j] = xn16[(size_t)fmess[m] * H + j];
}

// sum_h / sum_gh via neighbor gather; r-gate fused (r1b = x@Wr + Ur_b)
__global__ void gate_gather(const float* __restrict__ h32, const float* __restrict__ hUr,
                            const float* __restrict__ r1b, const int* __restrict__ mg,
                            float* __restrict__ sumh32, half_t* __restrict__ sumh16,
                            half_t* __restrict__ sumgh16) {
    int m = blockIdx.x, j = threadIdx.x;
    size_t base = (size_t)m * H + j;
    const int* g = mg + (size_t)m * MAXNEI;
    float rb = r1b[base];
    float s = 0.0f, gsum = 0.0f;
#pragma unroll
    for (int k = 0; k < MAXNEI; ++k) {
        size_t ib = (size_t)g[k] * H + j;
        float hv = h32[ib];
        float rv = sigmoidf_(rb + hUr[ib]);
        s += hv;
        gsum += rv * hv;
    }
    sumh32[base]  = s;
    sumh16[base]  = (_Float16)s;
    sumgh16[base] = (_Float16)gsum;
}

__global__ void node_gather(const float* __restrict__ h32, const int* __restrict__ ng,
                            half_t* __restrict__ mn16) {
    int n = blockIdx.x, j = threadIdx.x;
    const int* g = ng + (size_t)n * MAXNEI;
    float s = 0.0f;
#pragma unroll
    for (int k = 0; k < MAXNEI; ++k) s += h32[(size_t)g[k] * H + j];
    mn16[(size_t)n * H + j] = (_Float16)s;
}

__global__ void count_k(const int* __restrict__ scope, int* __restrict__ cnt, int n) {
    int i = blockIdx.x * blockDim.x + threadIdx.x;
    if (i < n) atomicAdd(&cnt[scope[i]], 1);
}

__global__ void finalize(const float* __restrict__ sums, const int* __restrict__ cnt,
                         float* __restrict__ out) {
    int t = blockIdx.x, j = threadIdx.x;
    float c = (float)cnt[t];
    if (c < 1.0f) c = 1.0f;
    out[(size_t)t * H + j] = sums[(size_t)t * H + j] / c;
}

// ---------------------------------------------------------------------------
extern "C" void kernel_launch(void* const* d_in, const int* in_sizes, int n_in,
                              void* d_out, int out_size, void* d_ws, size_t ws_size,
                              hipStream_t stream) {
    (void)in_sizes; (void)n_in; (void)out_size; (void)ws_size;

    const int*   fnode      = (const int*)d_in[0];
    const int*   fmess      = (const int*)d_in[1];
    const int*   node_graph = (const int*)d_in[2];
    const int*   mess_graph = (const int*)d_in[3];
    const int*   scope_ids  = (const int*)d_in[4];
    const float* emb        = (const float*)d_in[5];
    const float* Wz_w       = (const float*)d_in[6];
    const float* Wz_b       = (const float*)d_in[7];
    const float* Wr_w       = (const float*)d_in[8];
    const float* Ur_w       = (const float*)d_in[9];
    const float* Ur_b       = (const float*)d_in[10];
    const float* Wh_w       = (const float*)d_in[11];
    const float* Wh_b       = (const float*)d_in[12];
    const float* out_w      = (const float*)d_in[13];
    const float* out_b      = (const float*)d_in[14];
    float* out = (float*)d_out;

    char* ws = (char*)d_ws;
    size_t off = 0;
    auto alloc = [&](size_t bytes) -> char* {
        char* p = ws + off;
        off = (off + bytes + 255) & ~(size_t)255;
        return p;
    };

    const size_t WB = (size_t)H * H * sizeof(half_t);
    const size_t Mh = (size_t)MMSG * H;
    const size_t Nh = (size_t)NNODE * H;

    half_t* wR  = (half_t*)alloc(WB);
    half_t* wUr = (half_t*)alloc(WB);
    half_t* wZt = (half_t*)alloc(WB);
    half_t* wZb = (half_t*)alloc(WB);
    half_t* wHt = (half_t*)alloc(WB);
    half_t* wHb = (half_t*)alloc(WB);
    half_t* wOt = (half_t*)alloc(WB);
    half_t* wOb = (half_t*)alloc(WB);

    half_t* xn16    = (half_t*)alloc(Nh * sizeof(half_t));
    half_t* x16     = (half_t*)alloc(Mh * sizeof(half_t));
    float*  r1b     = (float*)alloc(Mh * sizeof(float));   // x@Wr + Ur_b
    float*  xWz     = (float*)alloc(Mh * sizeof(float));   // x@Wz_top + Wz_b
    float*  xWh     = (float*)alloc(Mh * sizeof(float));   // x@Wh_top + Wh_b
    float*  xo      = (float*)alloc(Nh * sizeof(float));   // x_node@out_top + out_b
    float*  h32     = (float*)alloc(Mh * sizeof(float));
    half_t* h16     = (half_t*)alloc(Mh * sizeof(half_t));
    float*  hUr     = (float*)alloc(Mh * sizeof(float));
    float*  sumh32  = (float*)alloc(Mh * sizeof(float));
    half_t* sumh16  = (half_t*)alloc(Mh * sizeof(half_t)); // reused as mess_nei16
    half_t* sumgh16 = (half_t*)alloc(Mh * sizeof(half_t));
    float*  sums    = (float*)alloc((size_t)NTREE * H * sizeof(float) + NTREE * sizeof(int));
    int*    cnt     = (int*)(sums + (size_t)NTREE * H);

    dim3 blk(H);
    const int gM = MMSG / 128, gN = NNODE / 128;

    reorder_w<<<128, 32, 0, stream>>>(Wr_w,          wR);
    reorder_w<<<128, 32, 0, stream>>>(Ur_w,          wUr);
    reorder_w<<<128, 32, 0, stream>>>(Wz_w,          wZt);
    reorder_w<<<128, 32, 0, stream>>>(Wz_w + H * H,  wZb);
    reorder_w<<<128, 32, 0, stream>>>(Wh_w,          wHt);
    reorder_w<<<128, 32, 0, stream>>>(Wh_w + H * H,  wHb);
    reorder_w<<<128, 32, 0, stream>>>(out_w,         wOt);
    reorder_w<<<128, 32, 0, stream>>>(out_w + H * H, wOb);

    embed_node<<<NNODE, blk, 0, stream>>>(fnode, emb, xn16);
    embed_mess<<<MMSG,  blk, 0, stream>>>(fmess, xn16, x16);

    // depth-invariant GEMMs
    gemm_k256<<<gM, 256, 0, stream>>>(x16,  wR,  Ur_b,  r1b);
    gemm_k256<<<gM, 256, 0, stream>>>(x16,  wZt, Wz_b,  xWz);
    gemm_k256<<<gM, 256, 0, stream>>>(x16,  wHt, Wh_b,  xWh);
    gemm_k256<<<gN, 256, 0, stream>>>(xn16, wOt, out_b, xo);

    zero_f<<<(unsigned)((Mh + 255) / 256), blk, 0, stream>>>(h32, (long)Mh);
    zero_f<<<(unsigned)((Mh / 2 + 255) / 256), blk, 0, stream>>>((float*)h16, (long)(Mh / 2));

    // GRU depth loop
    for (int d = 0; d < 6; ++d) {
        gemm_k256<<<gM, 256, 0, stream>>>(h16, wUr, nullptr, hUr);
        gate_gather<<<MMSG, blk, 0, stream>>>(h32, hUr, r1b, mess_graph,
                                              sumh32, sumh16, sumgh16);
        gemm2_gate_combine<<<gM, 256, 0, stream>>>(sumh16, sumgh16, wZb, wHb,
                                                   xWz, xWh, sumh32, h32, h16);
    }

    // readout
    node_gather<<<NNODE, blk, 0, stream>>>(h32, node_graph, sumh16);
    zero_f<<<(unsigned)(((size_t)NTREE * H + NTREE + 255) / 256), blk, 0, stream>>>(
        sums, (long)((size_t)NTREE * H + NTREE));
    gemm_out_accum<<<gN, 256, 0, stream>>>(sumh16, wOb, xo, scope_ids, sums);
    count_k<<<(NNODE + 255) / 256, blk, 0, stream>>>(scope_ids, cnt, NNODE);
    finalize<<<NTREE, blk, 0, stream>>>(sums, cnt, out);
}
